// Aggregate_85220741087337
// MI455X (gfx1250) — matrix operand order repaired and checked by reading, not compiled, not run on hardware
//
#include <hip/hip_runtime.h>
#include <hip/hip_bf16.h>

// ---------------------------------------------------------------------------
// Problem: B=64, S=512, F_IN=F_OUT=1024, ALPHA=0.5
//   y = x @ W^T   (32768 x 1024 x 1024 GEMM, bf16 WMMA / f32 accumulate)
//   then EMA scan over S per (b, o) channel, in place.
// out = [hk (64*1024 f32), hiddens (64*512*1024 f32)]
// ---------------------------------------------------------------------------
#define B_DIM   64
#define S_DIM   512
#define K_DIM   1024
#define N_DIM   1024
#define M_TOTAL (B_DIM * S_DIM)   // 32768

#define BLK_M 128
#define BLK_N 128
#define BLK_K 32
#define LDP   40                  // LDS row pitch (elements): conflict-free, keeps 16B align
#define KT_STEPS (K_DIM / BLK_K)  // 32

typedef __attribute__((ext_vector_type(16))) __bf16 v16bf;
typedef __attribute__((ext_vector_type(8)))  float  v8f;

struct __attribute__((aligned(16))) U4 { unsigned int x, y, z, w; };

union FragBF {
    U4    q[2];
    v16bf v;
};

// ---- f32 -> packed bf16x2: round-half-up then pack the two high halves.
// With v_perm_b32: 3 VALU ops per 2 floats (2 adds + 1 perm).
__device__ __forceinline__ unsigned int pack2(float a, float b) {
    unsigned int ua = __float_as_uint(a) + 0x8000u;   // round-half-up into bit 16
    unsigned int ub = __float_as_uint(b) + 0x8000u;
#if __has_builtin(__builtin_amdgcn_perm)
    // {S0=ub, S1=ua}; sel bytes 2,3 of ua (low half) and 6,7 -> bytes 2,3 of ub
    return __builtin_amdgcn_perm(ub, ua, 0x07060302u);
#else
    return (ua >> 16) | (ub & 0xffff0000u);
#endif
}

__device__ __forceinline__ U4 pack8(const float4& f0, const float4& f1) {
    U4 p;
    p.x = pack2(f0.x, f0.y); p.y = pack2(f0.z, f0.w);
    p.z = pack2(f1.x, f1.y); p.w = pack2(f1.z, f1.w);
    return p;
}

// ---------------------------------------------------------------------------
// GEMM: y[m, n] = sum_k x[m, k] * W[n, k]
// 128x128 block tile, 8 waves (2 M x 4 N), each wave owns 64x32 C
// double-buffered padded LDS, one barrier per K-step, prefetch k+2
// ---------------------------------------------------------------------------
__global__ __launch_bounds__(256)
void gemm_bf16_wmma(const float* __restrict__ x,
                    const float* __restrict__ W,
                    float* __restrict__ y)
{
    __shared__ unsigned short lA[2][BLK_M * LDP];   // 2 x 10 KB
    __shared__ unsigned short lB[2][BLK_N * LDP];   // 2 x 10 KB

    const int tid  = threadIdx.x;
    const int lane = tid & 31;
    const int wave = tid >> 5;          // 0..7
    const int wm   = wave >> 2;         // 0..1  (M sub-tile of 64)
    const int wn   = wave & 3;          // 0..3  (N sub-tile of 32)

    const int m0 = blockIdx.x * BLK_M;
    const int n0 = blockIdx.y * BLK_N;

    // staging: 2 threads per row, 16 consecutive floats each (A and B identical)
    const int srow  = tid >> 1;          // 0..127
    const int skoff = (tid & 1) * 16;    // 0 or 16

    const float* aSrc = x + (size_t)(m0 + srow) * K_DIM + skoff;
    const float* bSrc = W + (size_t)(n0 + srow) * K_DIM + skoff;

    // fragment lane mapping (ISA 7.12.2)
    const int lm  = lane & 15;
    const int ahk = (lane >> 4) * 8;     // A: k chunks {ahk..+7, 16+ahk..+7}
    const int bhk = (lane >> 4) * 8;     // B: k chunks {bhk..+7, 16+bhk..+7}

    v8f acc[4][2] = {};

    // prologue: load tile 0 into registers
    float4 ar[4], br[4];
    #pragma unroll
    for (int i = 0; i < 4; ++i) {
        ar[i] = ((const float4*)aSrc)[i];
        br[i] = ((const float4*)bSrc)[i];
    }

    int pp = 0;
    for (int kt = 0; kt < KT_STEPS; ++kt) {
        // ---- convert staged registers (tile kt) and store to LDS[pp] ----
        {
            U4* da = (U4*)&lA[pp][srow * LDP + skoff];
            da[0] = pack8(ar[0], ar[1]);
            da[1] = pack8(ar[2], ar[3]);
            U4* db = (U4*)&lB[pp][srow * LDP + skoff];
            db[0] = pack8(br[0], br[1]);
            db[1] = pack8(br[2], br[3]);
        }
        __syncthreads();

        // ---- prefetch tile kt+2 (L2), load tile kt+1 into registers ----
        if (kt + 2 < KT_STEPS) {
            __builtin_prefetch(aSrc + (size_t)(kt + 2) * BLK_K, 0, 3);
            __builtin_prefetch(bSrc + (size_t)(kt + 2) * BLK_K, 0, 3);
        }
        if (kt + 1 < KT_STEPS) {
            const float4* an = (const float4*)(aSrc + (size_t)(kt + 1) * BLK_K);
            const float4* bn = (const float4*)(bSrc + (size_t)(kt + 1) * BLK_K);
            #pragma unroll
            for (int i = 0; i < 4; ++i) { ar[i] = an[i]; br[i] = bn[i]; }
        }

        // ---- fragments from LDS[pp] ----
        FragBF a[4], b[2];
        #pragma unroll
        for (int mi = 0; mi < 4; ++mi) {
            const int row = wm * 64 + mi * 16 + lm;
            a[mi].q[0] = *(const U4*)&lA[pp][row * LDP + ahk];
            a[mi].q[1] = *(const U4*)&lA[pp][row * LDP + 16 + ahk];
        }
        #pragma unroll
        for (int ni = 0; ni < 2; ++ni) {
            const int nn = wn * 32 + ni * 16 + lm;
            b[ni].q[0] = *(const U4*)&lB[pp][nn * LDP + bhk];
            b[ni].q[1] = *(const U4*)&lB[pp][nn * LDP + bhk + 16];
        }

        // ---- 8x v_wmma_f32_16x16x32_bf16 ----
        #pragma unroll
        for (int mi = 0; mi < 4; ++mi)
            #pragma unroll
            for (int ni = 0; ni < 2; ++ni)
                acc[mi][ni] = __builtin_amdgcn_wmma_f32_16x16x32_bf16(
                    false, a[mi].v, false, b[ni].v,
                    (short)0, acc[mi][ni], false, false);

        pp ^= 1;
    }

    // ---- epilogue: C layout — lanes 0-15: M=r, lanes 16-31: M=8+r, N=lane&15
    const int mo = (lane >> 4) * 8;
    #pragma unroll
    for (int mi = 0; mi < 4; ++mi) {
        #pragma unroll
        for (int ni = 0; ni < 2; ++ni) {
            const int row = m0 + wm * 64 + mi * 16 + mo;
            const int col = n0 + wn * 32 + ni * 16 + lm;
            float* dst = y + (size_t)row * N_DIM + col;
            #pragma unroll
            for (int r = 0; r < 8; ++r)
                dst[(size_t)r * N_DIM] = acc[mi][ni][r];
        }
    }
}

// ---------------------------------------------------------------------------
// EMA scan over S, in place; one thread per (b, o) channel.
// ---------------------------------------------------------------------------
__global__ __launch_bounds__(256)
void ema_scan(float* __restrict__ hid, float* __restrict__ hk)
{
    const int idx = blockIdx.x * 256 + threadIdx.x;   // 0 .. B*N-1
    const int b = idx >> 10;
    const int o = idx & (N_DIM - 1);

    float* p = hid + (size_t)b * S_DIM * N_DIM + o;
    float h = 0.0f;
    for (int s = 0; s < S_DIM; ++s) {
        float v = p[(size_t)s * N_DIM];
        h = 0.5f * (h + v);
        p[(size_t)s * N_DIM] = h;
    }
    hk[idx] = h;
}

// ---------------------------------------------------------------------------
extern "C" void kernel_launch(void* const* d_in, const int* in_sizes, int n_in,
                              void* d_out, int out_size, void* d_ws, size_t ws_size,
                              hipStream_t stream)
{
    (void)in_sizes; (void)n_in; (void)out_size; (void)d_ws; (void)ws_size;

    const float* x = (const float*)d_in[0];   // [64, 512, 1024] f32
    const float* W = (const float*)d_in[1];   // [1024, 1024] f32

    float* out = (float*)d_out;
    float* hk  = out;                                  // [64, 1024]
    float* hid = out + (size_t)B_DIM * N_DIM;          // [64, 512, 1024]

    dim3 grid_gemm(M_TOTAL / BLK_M, N_DIM / BLK_N);    // (256, 8)
    gemm_bf16_wmma<<<grid_gemm, 256, 0, stream>>>(x, W, hid);

    ema_scan<<<(B_DIM * N_DIM) / 256, 256, 0, stream>>>(hid, hk);
}
